// IDMForwardSim_28166395527647
// MI455X (gfx1250) — matrix operand order repaired and checked by reading, not compiled, not run on hardware
//
#include <hip/hip_runtime.h>

typedef __attribute__((ext_vector_type(16))) _Float16 v16h;
typedef __attribute__((ext_vector_type(8)))  _Float16 v8h;
typedef __attribute__((ext_vector_type(8)))  float    v8f;

#define B_    8192
#define T_    40
#define H_    100
#define NCOL  448      // 4 gates * 112 packed columns
#define GS    112      // gate stride (7 tiles of 16)
#define KP    128      // padded K (100 h + 2 sdv + 26 zero)
#define ROWS  64       // batch rows per block
#define JT    7        // column tiles per gate

__device__ __forceinline__ float fsig(float x) {
  return __builtin_amdgcn_rcpf(1.0f + __expf(-x));
}
__device__ __forceinline__ float ftanh(float x) {
  float e = __expf(2.0f * x);
  return (e - 1.0f) * __builtin_amdgcn_rcpf(e + 1.0f);
}

__device__ __forceinline__ float idm_act(float vel, float dv, float dx,
                                         float inv_dv, float tgap, float mj,
                                         float mxa, float inv2s) {
  dx = fminf(fmaxf(dx, 0.5f), 1000.0f);
  float dg = mj + fmaxf(0.0f, tgap * vel + vel * dv * inv2s);
  float q = vel * inv_dv;
  float q2 = q * q;
  float r = dg * __builtin_amdgcn_rcpf(dx);
  float a = mxa * (1.0f - q2 * q2 - r * r);
  return fminf(fmaxf(a, -3.0f), 3.0f);
}

// ---------------------------------------------------------------------------
// Prep 1: pack [Wh(100x400); Wx[100:102](2x400)] into f16, gate-interleaved,
// column-major over K (wcat[col*KP + k]), col = g*112 + cc (cc<100 real).
// ---------------------------------------------------------------------------
__global__ void prep_wcat(const float* __restrict__ Wh, const float* __restrict__ Wx,
                          _Float16* __restrict__ wcat) {
  int idx = blockIdx.x * 256 + threadIdx.x;
  if (idx >= NCOL * KP) return;
  int col = idx >> 7;
  int k   = idx & (KP - 1);
  int g   = col / GS;
  int cc  = col % GS;
  float v = 0.0f;
  if (cc < H_) {
    int gcol = g * H_ + cc;
    if (k < H_)        v = Wh[k * 400 + gcol];
    else if (k < 102)  v = Wx[k * 400 + gcol];   // rows 100,101 of Wx (sdv part)
  }
  wcat[idx] = (_Float16)v;
}

// ---------------------------------------------------------------------------
// Prep 2 (one block per batch row):
//   att_proj = z @ linear_W + linear_b           (B x 100)
//   pre      = att_proj @ Wx[0:100,:] + lstm_b   (B x 448, gate-packed)
// ---------------------------------------------------------------------------
__global__ __launch_bounds__(128) void prep_row(
    const float* __restrict__ z, const float* __restrict__ lW,
    const float* __restrict__ lb, const float* __restrict__ Wx,
    const float* __restrict__ lstm_b, float* __restrict__ attproj,
    float* __restrict__ pre) {
  __shared__ float ap[H_];
  int row = blockIdx.x;
  int tid = threadIdx.x;
  if (tid < H_) {
    float s = lb[tid];
#pragma unroll
    for (int k = 0; k < 10; ++k) s += z[row * 10 + k] * lW[k * H_ + tid];
    ap[tid] = s;
    attproj[row * H_ + tid] = s;
  }
  __syncthreads();
  for (int pc = tid; pc < NCOL; pc += 128) {
    int g = pc / GS, cc = pc % GS;
    float s = 0.0f;
    if (cc < H_) {
      int gcol = g * H_ + cc;
      s = lstm_b[gcol];
      for (int k = 0; k < H_; ++k) s += ap[k] * Wx[k * 400 + gcol];
    }
    pre[row * NCOL + pc] = s;
  }
}

// ---------------------------------------------------------------------------
// Main recurrent kernel: 128 threads (4 waves), 64 batch rows per block.
// Each wave owns one 16-row tile; c-state in LDS (per-thread), h via LDS f16.
// ---------------------------------------------------------------------------
__global__ __launch_bounds__(128, 1) void idm_main(
    const float* __restrict__ idmp, const float* __restrict__ idms,
    const float* __restrict__ sdv, const float* __restrict__ attW,
    const float* __restrict__ attb, const float* __restrict__ noise_f,
    const float* __restrict__ noise_m, const float* __restrict__ attproj,
    const float* __restrict__ pre, const _Float16* __restrict__ wcat,
    float* __restrict__ out) {
  extern __shared__ char smemraw[];
  float*    sAttW = (float*)smemraw;                 // 101 floats (attW + att_b)
  _Float16* sWcat = (_Float16*)(smemraw + 512);      // 448*128 f16 = 112 KB
  _Float16* sA    = sWcat + NCOL * KP;               // 64*128 f16  = 16 KB
  float*    cLDS  = (float*)(sA + ROWS * KP);        // 128 thr * 56 f32 = 28 KB

  const int tid  = threadIdx.x;
  const int wave = tid >> 5;
  const int lane = tid & 31;
  const int n    = lane & 15;
  const bool hiHalf = lane >= 16;
  const int rowbase = blockIdx.x * ROWS;

  if (tid < H_) sAttW[tid] = attW[tid];
  if (tid == 0) sAttW[H_] = attb[0];

  // Weights -> LDS (16B vector copies)
  {
    const v8h* src = (const v8h*)wcat;
    v8h* dst = (v8h*)sWcat;
    for (int i = tid; i < NCOL * KP / 8; i += 128) dst[i] = src[i];
  }
  // h panel init: h0 = att_proj (K 100..127 zero; 100/101 get sdv each step)
  for (int i = tid; i < ROWS * KP; i += 128) {
    int r = i >> 7, k = i & (KP - 1);
    float v = (k < H_) ? attproj[(rowbase + r) * H_ + k] : 0.0f;
    sA[i] = (_Float16)v;
  }

  // c0 = att_proj in WMMA D layout, stored per-thread in LDS:
  // lane<16 -> M=vgpr, N=lane; lane>=16 -> M=vgpr+8, N=lane-16
  const int rowD0 = rowbase + wave * 16 + (hiHalf ? 8 : 0);
  {
    float* cp = cLDS + tid * (JT * 8);
#pragma unroll
    for (int j = 0; j < JT; ++j) {
      int k = j * 16 + n;
#pragma unroll
      for (int v = 0; v < 8; ++v)
        cp[j * 8 + v] = (k < H_) ? attproj[(rowD0 + v) * H_ + k] : 0.0f;
    }
  }

  // Per-row physics state (thread tid < ROWS owns row rowbase+tid)
  float ego_v = 0.f, ego_x = 0.f, act = 0.f;
  float p_invdv = 1.f, p_tg = 0.f, p_mj = 0.f, p_mxa = 0.f, p_inv2s = 0.f;
  const int prow = rowbase + tid;
  if (tid < ROWS) {
    ego_v = idms[prow * T_ * 8 + 0];
    ego_x = idms[prow * T_ * 8 + 3];
    p_invdv = 1.0f / idmp[prow * 5 + 0];
    p_tg    = idmp[prow * 5 + 1];
    p_mj    = idmp[prow * 5 + 2];
    p_mxa   = idmp[prow * 5 + 3];
    float mna = idmp[prow * 5 + 4];
    p_inv2s = 1.0f / (2.0f * sqrtf(p_mxa * mna));
  }

#pragma unroll 1
  for (int t = 0; t < T_; ++t) {
    // ---- phase 1: current sdv into A-panel K slots 100,101 ----
    if (tid < ROWS) {
      float2 sv = *(const float2*)(sdv + prow * T_ * 2 + t * 2);
      sA[tid * KP + 100] = (_Float16)sv.x;
      sA[tid * KP + 101] = (_Float16)sv.y;
    }
    __syncthreads();

    // ---- phase 2: z = pre + [h,sdv] @ Wcat via WMMA, then gates ----
    // A fragments (16x32 f16): lane<16 M=lane, K=i; lane>=16 K+=16
    v16h afr[4];
    {
      const _Float16* arow = sA + (wave * 16 + n) * KP;
#pragma unroll
      for (int kc = 0; kc < 4; ++kc) {
        int base = kc * 32 + (hiHalf ? 16 : 0);
        v8h lo = *(const v8h*)(arow + base);
        v8h hi = *(const v8h*)(arow + base + 8);
        afr[kc] = __builtin_shufflevector(lo, hi, 0, 1, 2, 3, 4, 5, 6, 7,
                                          8, 9, 10, 11, 12, 13, 14, 15);
      }
    }

#pragma unroll 1
    for (int j = 0; j < JT; ++j) {
      const int k = j * 16 + n;
      // acc init from pre: single base, all offsets compile-time immediates
      const float* pp = pre + (size_t)rowD0 * NCOL + k;
      v8f acc[4];
#pragma unroll
      for (int g = 0; g < 4; ++g)
#pragma unroll
        for (int v = 0; v < 8; ++v)
          acc[g][v] = pp[g * GS + (size_t)v * NCOL];

      // B fragment base (32x16 f16): lane%16 = N; lane<16 K=kc*32+i, else +16
      const _Float16* bbase = sWcat + (j * 16 + n) * KP + (hiHalf ? 16 : 0);
#pragma unroll
      for (int kc = 0; kc < 4; ++kc) {
        v16h bfr[4];
#pragma unroll
        for (int g = 0; g < 4; ++g)
          bfr[g] = *(const v16h*)(bbase + g * (JT * 16 * KP) + kc * 32);
#pragma unroll
        for (int g = 0; g < 4; ++g)
          acc[g] = __builtin_amdgcn_wmma_f32_16x16x32_f16(
              false, afr[kc], false, bfr[g], (short)0, acc[g], false, false);
      }

      // gate math in D layout; c in LDS, h -> LDS f16 panel
      float* cp = cLDS + tid * (JT * 8) + j * 8;
      v8f cv = *(const v8f*)cp;
      v8f cn;
      _Float16* hst = sA + (wave * 16 + (hiHalf ? 8 : 0)) * KP + k;
#pragma unroll
      for (int v = 0; v < 8; ++v) {
        float zi = acc[0][v], zf = acc[1][v], zg = acc[2][v], zo = acc[3][v];
        float c2 = fsig(zf) * cv[v] + fsig(zi) * ftanh(zg);
        cn[v] = c2;
        float hn = fsig(zo) * ftanh(c2);
        if (k < H_) hst[v * KP] = (_Float16)hn;
      }
      *(v8f*)cp = cn;
    }
    __syncthreads();

    // ---- phase 3: attention + IDM physics (one thread per row) ----
    if (tid < ROWS) {
      float dot = sAttW[H_];
      const _Float16* hr = sA + tid * KP;
#pragma unroll
      for (int c = 0; c < 12; ++c) {
        v8h hv = *(const v8h*)(hr + c * 8);
#pragma unroll
        for (int i = 0; i < 8; ++i) dot += (float)hv[i] * sAttW[c * 8 + i];
      }
#pragma unroll
      for (int kk = 96; kk < H_; ++kk) dot += (float)hr[kk] * sAttW[kk];
      float att = fsig(5.0f * dot);

      ego_v += act * 0.1f;
      ego_x += ego_v * 0.1f + act * 0.005f;

      const float* sp = idms + prow * T_ * 8 + t * 8;
      float4 s0 = *(const float4*)sp;
      float4 s1 = *(const float4*)(sp + 4);
      float nf = noise_f[t * B_ + prow];
      float nm = noise_m[t * B_ + prow];

      float ef = idm_act(ego_v, ego_v - s0.y, s1.x - ego_x,
                         p_invdv, p_tg, p_mj, p_mxa, p_inv2s);
      ef = s1.z * ef + (1.0f - s1.z) * nf;
      float em = idm_act(ego_v, ego_v - s0.z, s1.y - ego_x,
                         p_invdv, p_tg, p_mj, p_mxa, p_inv2s);
      em = s1.w * em + (1.0f - s1.w) * nm;

      act = (1.0f - att) * ef + att * em;
      out[prow * T_ + t] = act;
      out[B_ * T_ + prow * T_ + t] = att;
    }
  }
}

// ---------------------------------------------------------------------------
extern "C" void kernel_launch(void* const* d_in, const int* in_sizes, int n_in,
                              void* d_out, int out_size, void* d_ws, size_t ws_size,
                              hipStream_t stream) {
  const float* z      = (const float*)d_in[0];
  // d_in[1] enc_h: unused by the reference
  const float* idmp   = (const float*)d_in[2];
  const float* idms   = (const float*)d_in[3];
  const float* sdv    = (const float*)d_in[4];
  const float* lW     = (const float*)d_in[5];
  const float* lb     = (const float*)d_in[6];
  const float* Wx     = (const float*)d_in[7];
  const float* Wh     = (const float*)d_in[8];
  const float* lstm_b = (const float*)d_in[9];
  const float* attW   = (const float*)d_in[10];
  const float* attb   = (const float*)d_in[11];
  const float* nf     = (const float*)d_in[12];
  const float* nm     = (const float*)d_in[13];
  float* out = (float*)d_out;

  char* ws = (char*)d_ws;
  float*    attproj = (float*)ws;                                   //  8192*100 f32
  float*    pre     = (float*)(ws + (size_t)B_ * H_ * 4);           //  8192*448 f32
  _Float16* wcat    = (_Float16*)(ws + (size_t)B_ * H_ * 4 +
                                       (size_t)B_ * NCOL * 4);      //  448*128 f16

  prep_wcat<<<(NCOL * KP + 255) / 256, 256, 0, stream>>>(Wh, Wx, wcat);
  prep_row<<<B_, 128, 0, stream>>>(z, lW, lb, Wx, lstm_b, attproj, pre);

  size_t shmem = 512 + (size_t)NCOL * KP * 2 + (size_t)ROWS * KP * 2 +
                 (size_t)128 * JT * 8 * 4;                          // ~157 KB
  idm_main<<<B_ / ROWS, 128, shmem, stream>>>(idmp, idms, sdv, attW, attb,
                                              nf, nm, attproj, pre, wcat, out);
}
